// BERT_90469191123424
// MI455X (gfx1250) — matrix operand order repaired and checked by reading, not compiled, not run on hardware
//
#include <hip/hip_runtime.h>

// ---------------------------------------------------------------------------
// Types / helpers
// ---------------------------------------------------------------------------
typedef __attribute__((ext_vector_type(16))) __bf16 v16bf;
typedef __attribute__((ext_vector_type(8)))  float  v8f;

union Frag {                       // one WMMA 16-bf16 operand fragment (8 VGPRs)
  v16bf          v;
  unsigned short u[16];
};
static_assert(sizeof(Frag) == 32, "frag size");

__device__ __forceinline__ unsigned short f2b(float f) {  // fp32 -> bf16 RNE
  unsigned u = __builtin_bit_cast(unsigned, f);
  u += 0x7FFFu + ((u >> 16) & 1u);
  return (unsigned short)(u >> 16);
}

__device__ __forceinline__ void ld16(Frag& f, int half, const void* p) {
  reinterpret_cast<uint4*>(&f)[half] = *reinterpret_cast<const uint4*>(p);
}

__device__ __forceinline__ v8f wmma_bf16(const Frag& a, const Frag& b, v8f c) {
  return __builtin_amdgcn_wmma_f32_16x16x32_bf16(false, a.v, false, b.v,
                                                 (short)0, c, false, false);
}

__device__ __forceinline__ v8f vzero8() {
  v8f z;
#pragma unroll
  for (int i = 0; i < 8; ++i) z[i] = 0.f;
  return z;
}

__device__ __forceinline__ float block_reduce(float v, float* red) {
  int t = threadIdx.x;
  red[t] = v;
  __syncthreads();
#pragma unroll
  for (int off = 128; off > 0; off >>= 1) {
    if (t < off) red[t] += red[t + off];
    __syncthreads();
  }
  float r = red[0];
  __syncthreads();
  return r;
}

// ---------------------------------------------------------------------------
// Problem constants
// ---------------------------------------------------------------------------
#define BB  16
#define SS  512
#define DD  768
#define HH  12
#define LL  4
#define FF_ 3072
#define MM  (BB * SS)   // 8192 rows
#define DHH 64

// ---------------------------------------------------------------------------
// fp32 -> bf16 convert WITH transpose: src [R,C] f32 -> dst [C,R] bf16
// (per-matrix via blockIdx.z), tiled 32x32 through LDS
// ---------------------------------------------------------------------------
__global__ __launch_bounds__(256) void cvt_t_kernel(const float* __restrict__ src,
                                                    unsigned short* __restrict__ dst,
                                                    int R, int C) {
  __shared__ unsigned short t[32][33];
  const int z = blockIdx.z;
  const float* s = src + (size_t)z * R * C;
  unsigned short* d = dst + (size_t)z * R * C;
  const int r0 = blockIdx.y * 32, c0 = blockIdx.x * 32;
  const int tx = threadIdx.x & 31, ty = threadIdx.x >> 5;   // 32 x 8
#pragma unroll
  for (int i = 0; i < 4; ++i) {
    int rr = ty + i * 8;
    t[rr][tx] = f2b(s[(size_t)(r0 + rr) * C + c0 + tx]);
  }
  __syncthreads();
#pragma unroll
  for (int i = 0; i < 4; ++i) {
    int cc = ty + i * 8;
    d[(size_t)(c0 + cc) * R + r0 + tx] = t[tx][cc];
  }
}

// ---------------------------------------------------------------------------
// Embedding + LayerNorm (eps 1e-12): one block per token row
// ---------------------------------------------------------------------------
__global__ __launch_bounds__(256) void embed_ln_kernel(
    const float* __restrict__ we, const float* __restrict__ pe,
    const float* __restrict__ te, const float* __restrict__ g,
    const float* __restrict__ bta, const int* __restrict__ ids,
    const int* __restrict__ tts, float* __restrict__ xf,
    unsigned short* __restrict__ xb) {
  __shared__ float red[256];
  int bs = blockIdx.x;
  int s  = bs % SS;
  int id = ids[bs], tt = tts[bs];
  int t  = threadIdx.x;
  float vals[3];
  float lsum = 0.f;
#pragma unroll
  for (int j = 0; j < 3; ++j) {
    int c = t + j * 256;
    float e = we[(size_t)id * DD + c] + pe[(size_t)s * DD + c] + te[(size_t)tt * DD + c];
    vals[j] = e;
    lsum += e;
  }
  float mean = block_reduce(lsum, red) * (1.f / DD);
  float lv = 0.f;
#pragma unroll
  for (int j = 0; j < 3; ++j) { float d = vals[j] - mean; lv += d * d; }
  float var = block_reduce(lv, red) * (1.f / DD);
  float inv = rsqrtf(var + 1e-12f);
#pragma unroll
  for (int j = 0; j < 3; ++j) {
    int c = t + j * 256;
    float y = (vals[j] - mean) * inv * g[c] + bta[c];
    xf[(size_t)bs * DD + c] = y;
    xb[(size_t)bs * DD + c] = f2b(y);
  }
}

// ---------------------------------------------------------------------------
// Residual add + LayerNorm (eps 1e-6): one block per row; in-place safe
// ---------------------------------------------------------------------------
__global__ __launch_bounds__(256) void add_ln_kernel(
    const float* __restrict__ xin, const float* __restrict__ oin,
    const float* __restrict__ g, const float* __restrict__ bta,
    float* __restrict__ xf, unsigned short* __restrict__ xb) {
  __shared__ float red[256];
  int bs = blockIdx.x;
  int t  = threadIdx.x;
  float vals[3];
  float lsum = 0.f;
#pragma unroll
  for (int j = 0; j < 3; ++j) {
    int c = t + j * 256;
    float e = xin[(size_t)bs * DD + c] + oin[(size_t)bs * DD + c];
    vals[j] = e;
    lsum += e;
  }
  float mean = block_reduce(lsum, red) * (1.f / DD);
  float lv = 0.f;
#pragma unroll
  for (int j = 0; j < 3; ++j) { float d = vals[j] - mean; lv += d * d; }
  float var = block_reduce(lv, red) * (1.f / DD);
  float inv = rsqrtf(var + 1e-6f);
#pragma unroll
  for (int j = 0; j < 3; ++j) {
    int c = t + j * 256;
    float y = (vals[j] - mean) * inv * g[c] + bta[c];
    xf[(size_t)bs * DD + c] = y;
    xb[(size_t)bs * DD + c] = f2b(y);
  }
}

// ---------------------------------------------------------------------------
// Prefix-mask key counts
// ---------------------------------------------------------------------------
__global__ __launch_bounds__(256) void count_kernel(const int* __restrict__ mask,
                                                    int* __restrict__ counts) {
  __shared__ int r[256];
  int b = blockIdx.x, t = threadIdx.x;
  int c = (mask[b * SS + t] != 0) + (mask[b * SS + 256 + t] != 0);
  r[t] = c;
  __syncthreads();
#pragma unroll
  for (int off = 128; off > 0; off >>= 1) {
    if (t < off) r[t] += r[t + off];
    __syncthreads();
  }
  if (t == 0) counts[b] = r[0];
}

// ---------------------------------------------------------------------------
// WMMA GEMM: C[M,N] = A[M,K] (bf16) * Wt[N,K] (bf16, pre-transposed) + bias.
// Block tile 128x128, 8 waves (4M x 2N), wave tile 32x64 (2x4 WMMA frags).
// Double-buffered LDS staging via GLOBAL_LOAD_ASYNC_TO_LDS_B128 (ASYNCcnt),
// one barrier per K-step.
// ---------------------------------------------------------------------------
#define GBM 128
#define GBN 128
#define GBK 32
#define LDA 40   // bf16 units: 80-byte rows -> bank-conflict free
#define LDB 40
#define ABUF (GBM * LDA)   // 5120 u16 = 10240 B per buffer
#define BBUF (GBN * LDB)

template <bool RELU, bool OF32, bool OBF>
__global__ __launch_bounds__(256) void gemm_kernel(
    const unsigned short* __restrict__ A, const unsigned short* __restrict__ Wt,
    const float* __restrict__ bias, float* __restrict__ outf,
    unsigned short* __restrict__ outb, int M, int N, int K) {
  __shared__ unsigned short As[2 * ABUF];
  __shared__ unsigned short Bs[2 * BBUF];

  const int tid  = threadIdx.x;
  const int lane = tid & 31, wave = tid >> 5;
  const int wm = wave >> 1, wn = wave & 1;
  const int hi = lane >> 4, ln = lane & 15;
  const int m0 = blockIdx.y * GBM, n0 = blockIdx.x * GBN;

  // raw LDS byte addresses (addrspace(3) offset lives in the low 32 bits)
  const unsigned asb = (unsigned)(size_t)(&As[0]);
  const unsigned bsb = (unsigned)(size_t)(&Bs[0]);

  // async global->LDS stage of one 128x32 A tile and one 128x32 Bt tile
  auto stage = [&](int buf, int k0) {
    unsigned ab  = asb + (unsigned)buf * (ABUF * 2);
    unsigned bb2 = bsb + (unsigned)buf * (BBUF * 2);
#pragma unroll
    for (int j = 0; j < 2; ++j) {
      int c = tid + 256 * j;                 // 512 chunks of 8 bf16
      int r = c >> 2, kc = (c & 3) << 3;
      const unsigned short* gp = A + (size_t)(m0 + r) * K + k0 + kc;
      unsigned la = ab + (unsigned)(r * LDA + kc) * 2;
      asm volatile("global_load_async_to_lds_b128 %0, %1, off"
                   :: "v"(la), "v"(gp) : "memory");
    }
#pragma unroll
    for (int j = 0; j < 2; ++j) {
      int c = tid + 256 * j;
      int n = c >> 2, kc = (c & 3) << 3;
      const unsigned short* gp = Wt + (size_t)(n0 + n) * K + k0 + kc;
      unsigned la = bb2 + (unsigned)(n * LDB + kc) * 2;
      asm volatile("global_load_async_to_lds_b128 %0, %1, off"
                   :: "v"(la), "v"(gp) : "memory");
    }
  };

  v8f acc[2][4];
#pragma unroll
  for (int i = 0; i < 2; ++i)
#pragma unroll
    for (int j = 0; j < 4; ++j) acc[i][j] = vzero8();

  const int nk = K / GBK;
  stage(0, 0);
  for (int s = 0; s < nk; ++s) {
    asm volatile("s_wait_asynccnt 0x0" ::: "memory");  // my tile writes landed
    __syncthreads();                                   // ...published to all waves
    if (s + 1 < nk) stage((s + 1) & 1, (s + 1) * GBK); // prefetch next tile

    const unsigned short* as = &As[(s & 1) * ABUF];
    const unsigned short* bs = &Bs[(s & 1) * BBUF];

    Frag af[2];
#pragma unroll
    for (int mt = 0; mt < 2; ++mt) {
      int row = wm * 32 + mt * 16 + ln;      // A frag: k = i%8 + (i/8)*16 + hi*8
      ld16(af[mt], 0, &as[row * LDA + hi * 8]);
      ld16(af[mt], 1, &as[row * LDA + 16 + hi * 8]);
    }
#pragma unroll
    for (int nt = 0; nt < 4; ++nt) {
      Frag bf_;
      int col = wn * 64 + nt * 16 + ln;      // B frag: k = i%8 + (i/8)*16 + hi*8
      ld16(bf_, 0, &bs[col * LDB + hi * 8]);
      ld16(bf_, 1, &bs[col * LDB + 16 + hi * 8]);
#pragma unroll
      for (int mt = 0; mt < 2; ++mt)
        acc[mt][nt] = wmma_bf16(af[mt], bf_, acc[mt][nt]);
    }
  }

#pragma unroll
  for (int mt = 0; mt < 2; ++mt)
#pragma unroll
    for (int nt = 0; nt < 4; ++nt) {
      int col = n0 + wn * 64 + nt * 16 + ln;
      float bv = bias[col];
#pragma unroll
      for (int r = 0; r < 8; ++r) {          // C frag: M = r + 8*hi, N = ln
        int row = m0 + wm * 32 + mt * 16 + r + hi * 8;
        float v = acc[mt][nt][r] + bv;
        if (RELU) v = v > 0.f ? v : 0.f;
        if (OF32) outf[(size_t)row * N + col] = v;
        if (OBF)  outb[(size_t)row * N + col] = f2b(v);
      }
    }
}

// ---------------------------------------------------------------------------
// Fused single-head attention: one wave handles one (b, h, 16-query tile).
// Scores via WMMA -> LDS, prefix mask + softmax, P*V via WMMA.
// ---------------------------------------------------------------------------
#define SCP 513   // scores f32 row stride (pad: conflict-free row reads)
#define PLP 520   // P bf16 row stride (1040B rows, 16B aligned)
#define VLP 72    // V tile bf16 row stride (144B rows)

__global__ __launch_bounds__(32) void attn_kernel(
    const unsigned short* __restrict__ q, const unsigned short* __restrict__ k,
    const unsigned short* __restrict__ v, const int* __restrict__ counts,
    unsigned short* __restrict__ out) {
  __shared__ float          sc[16 * SCP];    // 32832 B
  __shared__ unsigned short pl[16 * PLP];    // 16640 B
  __shared__ unsigned short vl[32 * VLP];    //  4608 B

  const int lane = threadIdx.x;
  const int hi = lane >> 4, ln = lane & 15;
  const int qt = blockIdx.x, h = blockIdx.y, b = blockIdx.z;
  const int cnt = counts[b];
  const size_t base = (size_t)b * SS * DD + (size_t)h * DHH;

  // Q fragments (16 queries x 64 dims = 2 K-steps)
  Frag aq[2];
  {
    size_t qrow = base + (size_t)(qt * 16 + ln) * DD;
#pragma unroll
    for (int c = 0; c < 2; ++c) {
      ld16(aq[c], 0, q + qrow + c * 32 + hi * 8);
      ld16(aq[c], 1, q + qrow + c * 32 + 16 + hi * 8);
    }
  }

  // ---- scores: 32 key tiles of 16
  for (int kt = 0; kt < 32; ++kt) {
    Frag bk[2];
    size_t krow = base + (size_t)(kt * 16 + ln) * DD;   // B frag: n=key, k=dim
#pragma unroll
    for (int c = 0; c < 2; ++c) {
      ld16(bk[c], 0, k + krow + c * 32 + hi * 8);
      ld16(bk[c], 1, k + krow + c * 32 + 16 + hi * 8);
    }
    v8f a = vzero8();
    a = wmma_bf16(aq[0], bk[0], a);
    a = wmma_bf16(aq[1], bk[1], a);
    int j = kt * 16 + ln;
#pragma unroll
    for (int r = 0; r < 8; ++r) {
      int m = r + hi * 8;
      float sv = a[r] * 0.125f;                 // 1/sqrt(64)
      if (j >= cnt) sv = -__builtin_inff();     // prefix key mask
      sc[m * SCP + j] = sv;
    }
  }
  __syncthreads();

  // ---- softmax: lane (ln, hi) handles row ln, half hi (256 cols)
  {
    int m = ln, j0 = hi * 256;
    float mx = -__builtin_inff();
    for (int j = j0; j < j0 + 256; ++j) mx = fmaxf(mx, sc[m * SCP + j]);
    mx = fmaxf(mx, __shfl_xor(mx, 16, 32));
    float sum = 0.f;
    for (int j = j0; j < j0 + 256; ++j) sum += __expf(sc[m * SCP + j] - mx);
    sum += __shfl_xor(sum, 16, 32);
    float inv = 1.f / sum;
    for (int j = j0; j < j0 + 256; ++j)
      pl[m * PLP + j] = f2b(__expf(sc[m * SCP + j] - mx) * inv);
  }
  __syncthreads();

  // ---- O = P * V : 16 key tiles of 32
  v8f o[4];
#pragma unroll
  for (int i = 0; i < 4; ++i) o[i] = vzero8();

  for (int kt = 0; kt < 16; ++kt) {
    // stage V tile (32 keys x 32 dims) into LDS: lane loads 32 dims of one key row
    {
      size_t vrow = base + (size_t)(kt * 32 + lane) * DD;
#pragma unroll
      for (int c = 0; c < 4; ++c)
        *reinterpret_cast<uint4*>(&vl[lane * VLP + c * 8]) =
            *reinterpret_cast<const uint4*>(v + vrow + c * 8);
    }
    __syncthreads();

    Frag pa;                                    // A frag of P (16 x 32 keys)
    ld16(pa, 0, &pl[ln * PLP + kt * 32 + hi * 8]);
    ld16(pa, 1, &pl[ln * PLP + kt * 32 + 16 + hi * 8]);
#pragma unroll
    for (int ng = 0; ng < 4; ++ng) {
      Frag vb;                                  // B frag: n=dim, k=key offset
#pragma unroll
      for (int i = 0; i < 16; ++i)
        vb.u[i] = vl[(hi * 8 + i + (i >> 3) * 8) * VLP + ng * 16 + ln];
      o[ng] = wmma_bf16(pa, vb, o[ng]);
    }
    __syncthreads();
  }

  // store attention output as bf16 [B*S, D] (col = h*64 + d)
#pragma unroll
  for (int ng = 0; ng < 4; ++ng)
#pragma unroll
    for (int r = 0; r < 8; ++r) {
      int m = r + hi * 8;
      out[(size_t)(b * SS + qt * 16 + m) * DD + h * DHH + ng * 16 + ln] =
          f2b(o[ng][r]);
    }
}

// ---------------------------------------------------------------------------
// CLS classifier + 2-way softmax
// ---------------------------------------------------------------------------
__global__ __launch_bounds__(256) void cls_kernel(const float* __restrict__ x,
                                                  const float* __restrict__ wf,
                                                  const float* __restrict__ bfin,
                                                  float* __restrict__ out) {
  __shared__ float red[256];
  int b = blockIdx.x, t = threadIdx.x;
  float s0 = 0.f, s1 = 0.f;
#pragma unroll
  for (int j = 0; j < 3; ++j) {
    int c = t + j * 256;
    float xv = x[(size_t)b * SS * DD + c];     // CLS token, s = 0
    s0 += xv * wf[c * 2 + 0];
    s1 += xv * wf[c * 2 + 1];
  }
  float l0 = block_reduce(s0, red);
  float l1 = block_reduce(s1, red);
  if (t == 0) {
    l0 += bfin[0];
    l1 += bfin[1];
    float mx = fmaxf(l0, l1);
    float e0 = __expf(l0 - mx), e1 = __expf(l1 - mx);
    float inv = 1.f / (e0 + e1);
    out[b * 2 + 0] = e0 * inv;
    out[b * 2 + 1] = e1 * inv;
  }
}

// ---------------------------------------------------------------------------
// Host orchestration
// ---------------------------------------------------------------------------
extern "C" void kernel_launch(void* const* d_in, const int* in_sizes, int n_in,
                              void* d_out, int out_size, void* d_ws, size_t ws_size,
                              hipStream_t stream) {
  const float* word_emb = (const float*)d_in[0];
  const float* pos_emb  = (const float*)d_in[1];
  const float* tok_emb  = (const float*)d_in[2];
  const float* emb_g    = (const float*)d_in[3];
  const float* emb_b    = (const float*)d_in[4];
  const float* Wq = (const float*)d_in[5];  const float* bq = (const float*)d_in[6];
  const float* Wk = (const float*)d_in[7];  const float* bk = (const float*)d_in[8];
  const float* Wv = (const float*)d_in[9];  const float* bv = (const float*)d_in[10];
  const float* Wo = (const float*)d_in[11]; const float* bo = (const float*)d_in[12];
  const float* W1 = (const float*)d_in[13]; const float* b1 = (const float*)d_in[14];
  const float* W2 = (const float*)d_in[15]; const float* b2 = (const float*)d_in[16];
  const float* ln_g = (const float*)d_in[17];
  const float* ln_b = (const float*)d_in[18];
  const float* Wf = (const float*)d_in[19]; const float* bfin = (const float*)d_in[20];
  const int* input_ids      = (const int*)d_in[21];
  const int* token_type_ids = (const int*)d_in[22];
  const int* mask           = (const int*)d_in[23];

  char* ws = (char*)d_ws;
  size_t off = 0;
  auto alloc = [&](size_t bytes) -> void* {
    void* p = ws + off;
    off += (bytes + 255) & ~(size_t)255;
    return p;
  };

  const size_t nDD  = (size_t)LL * DD * DD;   // per-projection weight elems (all L)
  const size_t nDFF = (size_t)LL * DD * FF_;

  unsigned short* wq_t = (unsigned short*)alloc(nDD * 2);   // [N=D, K=D] per layer
  unsigned short* wk_t = (unsigned short*)alloc(nDD * 2);
  unsigned short* wv_t = (unsigned short*)alloc(nDD * 2);
  unsigned short* wo_t = (unsigned short*)alloc(nDD * 2);
  unsigned short* w1_t = (unsigned short*)alloc(nDFF * 2);  // [N=FF, K=D]
  unsigned short* w2_t = (unsigned short*)alloc(nDFF * 2);  // [N=D, K=FF]

  float*          x_f  = (float*)alloc((size_t)MM * DD * 4);
  unsigned short* x_b  = (unsigned short*)alloc((size_t)MM * DD * 2);
  unsigned short* q_b  = (unsigned short*)alloc((size_t)MM * DD * 2);
  unsigned short* k_b  = (unsigned short*)alloc((size_t)MM * DD * 2);
  unsigned short* v_b  = (unsigned short*)alloc((size_t)MM * DD * 2);
  unsigned short* at_b = (unsigned short*)alloc((size_t)MM * DD * 2);
  float*          o_f  = (float*)alloc((size_t)MM * DD * 4);          // reused for FFN out
  unsigned short* h_b  = (unsigned short*)alloc((size_t)MM * FF_ * 2);
  int*            cnts = (int*)alloc(BB * sizeof(int));

  // 1) weight conversion fp32 -> bf16 WITH transpose ([K,N] -> [N,K])
  {
    dim3 gDxD(DD / 32, DD / 32, LL);
    cvt_t_kernel<<<gDxD, 256, 0, stream>>>(Wq, wq_t, DD, DD);
    cvt_t_kernel<<<gDxD, 256, 0, stream>>>(Wk, wk_t, DD, DD);
    cvt_t_kernel<<<gDxD, 256, 0, stream>>>(Wv, wv_t, DD, DD);
    cvt_t_kernel<<<gDxD, 256, 0, stream>>>(Wo, wo_t, DD, DD);
    dim3 gDxF(FF_ / 32, DD / 32, LL);   // W1 [D,FF] -> [FF,D]
    cvt_t_kernel<<<gDxF, 256, 0, stream>>>(W1, w1_t, DD, FF_);
    dim3 gFxD(DD / 32, FF_ / 32, LL);   // W2 [FF,D] -> [D,FF]
    cvt_t_kernel<<<gFxD, 256, 0, stream>>>(W2, w2_t, FF_, DD);
  }

  // 2) embeddings + LN, key counts
  embed_ln_kernel<<<MM, 256, 0, stream>>>(word_emb, pos_emb, tok_emb, emb_g, emb_b,
                                          input_ids, token_type_ids, x_f, x_b);
  count_kernel<<<BB, 256, 0, stream>>>(mask, cnts);

  const dim3 gD(DD / GBN, MM / GBM);    // (6, 64)
  const dim3 gF(FF_ / GBN, MM / GBM);   // (24, 64)
  const dim3 gA(SS / 16, HH, BB);       // (32, 12, 16)

  // 3) encoder layers
  for (int i = 0; i < LL; ++i) {
    const size_t wDD  = (size_t)i * DD * DD;
    const size_t wDFF = (size_t)i * DD * FF_;

    gemm_kernel<false, false, true><<<gD, 256, 0, stream>>>(
        x_b, wq_t + wDD, bq + i * DD, nullptr, q_b, MM, DD, DD);
    gemm_kernel<false, false, true><<<gD, 256, 0, stream>>>(
        x_b, wk_t + wDD, bk + i * DD, nullptr, k_b, MM, DD, DD);
    gemm_kernel<false, false, true><<<gD, 256, 0, stream>>>(
        x_b, wv_t + wDD, bv + i * DD, nullptr, v_b, MM, DD, DD);

    attn_kernel<<<gA, 32, 0, stream>>>(q_b, k_b, v_b, cnts, at_b);

    gemm_kernel<false, true, false><<<gD, 256, 0, stream>>>(
        at_b, wo_t + wDD, bo + i * DD, o_f, nullptr, MM, DD, DD);
    add_ln_kernel<<<MM, 256, 0, stream>>>(x_f, o_f, ln_g + i * DD, ln_b + i * DD,
                                          x_f, x_b);

    gemm_kernel<true, false, true><<<gF, 256, 0, stream>>>(
        x_b, w1_t + wDFF, b1 + i * FF_, nullptr, h_b, MM, FF_, DD);
    gemm_kernel<false, true, false><<<gD, 256, 0, stream>>>(
        h_b, w2_t + wDFF, b2 + i * DD, o_f, nullptr, MM, DD, FF_);
    add_ln_kernel<<<MM, 256, 0, stream>>>(x_f, o_f, ln_g + i * DD, ln_b + i * DD,
                                          x_f, x_b);
  }

  // 4) classifier
  cls_kernel<<<BB, 256, 0, stream>>>(x_f, Wf, bfin, (float*)d_out);

  (void)in_sizes; (void)n_in; (void)out_size; (void)ws_size;
}
